// PointNetPPFeatureExtractor_64647847740099
// MI455X (gfx1250) — matrix operand order repaired and checked by reading, not compiled, not run on hardware
//
#include <hip/hip_runtime.h>

// ---------------------------------------------------------------------------
// DGCNN EdgeConv feature extractor for MI455X (gfx1250, wave32, WMMA).
// f16 activations, f32 WMMA accumulation, f32 BN statistics.
// GEMMs use double-buffered LDS tiles filled by GLOBAL_LOAD_ASYNC_TO_LDS_B128
// (ASYNCcnt-tracked, no VGPR staging), WMMA 16x16x32 f16 compute.
// ---------------------------------------------------------------------------

typedef _Float16 h8   __attribute__((ext_vector_type(8)));
typedef _Float16 v16h __attribute__((ext_vector_type(16)));
typedef float    v8f  __attribute__((ext_vector_type(8)));

#define BN_EPS 1e-5f
#define KNN 20
#define NB 8
#define NPTS 4096
#define NEDGE (NB * NPTS * KNN)   // 655360
#define NPOINT (NB * NPTS)        // 32768

// async copy: 16B per lane, cache -> LDS, tracked by ASYNCcnt.
// lds32: LDS byte address (low 32 bits of generic pointer), off32: byte offset
// from the uniform SGPR base.  (GVS addressing: SGPR_U64 + VGPR_I32.)
#define ASYNC_CP16(lds32, off32, base)                                        \
    asm volatile("global_load_async_to_lds_b128 %0, %1, %2"                   \
                 :: "v"(lds32), "v"(off32), "s"(base) : "memory")

// ---------------------------------------------------------------------------
// KNN + edge feature + layer-1 (6 -> 64).  One thread per point; the whole
// batch's 4096 points live in LDS (48 KB of the WGP's 320 KB).
// ---------------------------------------------------------------------------
__global__ __launch_bounds__(256) void knn_edge_l1(
    const float* __restrict__ x, const float* __restrict__ w1,
    const float* __restrict__ b1, _Float16* __restrict__ P1)
{
    __shared__ float sx[NPTS], sy[NPTS], sz[NPTS];
    __shared__ float sw1[64 * 6];
    __shared__ float sb1[64];

    const int tid = threadIdx.x;
    const int b   = blockIdx.y;
    const float* xb = x + (size_t)b * NPTS * 3;
    for (int i = tid; i < NPTS; i += 256) {
        sx[i] = xb[i * 3 + 0];
        sy[i] = xb[i * 3 + 1];
        sz[i] = xb[i * 3 + 2];
    }
    for (int i = tid; i < 64 * 6; i += 256) sw1[i] = w1[i];
    if (tid < 64) sb1[tid] = b1[tid];
    __syncthreads();

    const int n = blockIdx.x * 256 + tid;
    const float cx = sx[n], cy = sy[n], cz = sz[n];

    // top-20 smallest squared distances (== top_k of -d^2), register-resident
    float bd[KNN]; int bi[KNN];
    #pragma unroll
    for (int j = 0; j < KNN; ++j) { bd[j] = 3.0e38f; bi[j] = 0; }

    for (int m = 0; m < NPTS; ++m) {
        const float dx = sx[m] - cx;
        const float dy = sy[m] - cy;
        const float dz = sz[m] - cz;
        const float d2 = dx * dx + dy * dy + dz * dz;
        if (d2 < bd[KNN - 1]) {
            float dv = d2; int iv = m;
            #pragma unroll
            for (int j = 0; j < KNN; ++j) {
                if (dv < bd[j]) {
                    float tf = bd[j]; bd[j] = dv; dv = tf;
                    int   ti = bi[j]; bi[j] = iv; iv = ti;
                }
            }
        }
    }

    // edge = [nbr-ctr, ctr]; h1[o] = b1[o] + sum_c e[c]*w1[o,c]  (K=6: VALU)
    const size_t pbase = ((size_t)b * NPTS + n) * KNN;
    #pragma unroll
    for (int k = 0; k < KNN; ++k) {
        const int mm = bi[k];
        const float e0 = sx[mm] - cx;
        const float e1 = sy[mm] - cy;
        const float e2 = sz[mm] - cz;
        _Float16* out = P1 + (pbase + k) * 64;
        for (int o = 0; o < 64; ++o) {
            const float* wr = &sw1[o * 6];
            const float h = sb1[o] + e0 * wr[0] + e1 * wr[1] + e2 * wr[2]
                          + cx * wr[3] + cy * wr[4] + cz * wr[5];
            out[o] = (_Float16)h;
        }
    }
}

// ---------------------------------------------------------------------------
// Per-channel sum / sum-of-squares over rows of a [M, Nc] f16 matrix.
// ---------------------------------------------------------------------------
__global__ __launch_bounds__(256) void col_stats(
    const _Float16* __restrict__ P, float* __restrict__ gS,
    float* __restrict__ gQ, int M, int Nc)
{
    __shared__ float sS[256], sQ[256];
    const int tid = threadIdx.x;
    const int c   = tid % Nc;
    const int g   = tid / Nc;
    const int grp = 256 / Nc;
    float s = 0.0f, q = 0.0f;
    for (int row = blockIdx.x * grp + g; row < M; row += gridDim.x * grp) {
        const float v = (float)P[(size_t)row * Nc + c];
        s += v; q += v * v;
    }
    sS[tid] = s; sQ[tid] = q;
    __syncthreads();
    if (tid < Nc) {
        s = sS[tid]; q = sQ[tid];
        for (int gg = 1; gg < grp; ++gg) { s += sS[gg * Nc + tid]; q += sQ[gg * Nc + tid]; }
        atomicAdd(&gS[tid], s);
        atomicAdd(&gQ[tid], q);
    }
}

// ---------------------------------------------------------------------------
// stats layout per layer L (base = stats + L*1024 floats):
//   [0..255] sum, [256..511] sumsq, [512..767] scale, [768..1023] shift
// ---------------------------------------------------------------------------
__global__ void finalize_bn(float* __restrict__ S, const float* __restrict__ g,
                            const float* __restrict__ be, float cnt, int Nc)
{
    const int c = threadIdx.x;
    if (c < Nc) {
        const float mean = S[c] / cnt;
        const float var  = S[256 + c] / cnt - mean * mean;   // biased, like torch BN
        const float sc   = g[c] * rsqrtf(var + BN_EPS);
        S[512 + c] = sc;
        S[768 + c] = be[c] - mean * sc;
    }
}

// flat BN+ReLU, Nc must be a power of two (64 / 128 here)
__global__ __launch_bounds__(256) void bn_relu(
    _Float16* __restrict__ P, const float* __restrict__ scale,
    const float* __restrict__ shift, int ncMask, size_t total)
{
    const size_t stride = (size_t)gridDim.x * 256;
    for (size_t i = (size_t)blockIdx.x * 256 + threadIdx.x; i < total; i += stride) {
        const int c = (int)i & ncMask;
        const float v = (float)P[i] * scale[c] + shift[c];
        P[i] = (_Float16)fmaxf(v, 0.0f);
    }
}

__global__ __launch_bounds__(256) void bn_relu_maxk(
    const _Float16* __restrict__ P3, const float* __restrict__ scale,
    const float* __restrict__ shift, _Float16* __restrict__ A4)
{
    const int c = threadIdx.x;
    const size_t p = blockIdx.x;
    const float sc = scale[c], sh = shift[c];
    float m = -3.0e38f;
    for (int k = 0; k < KNN; ++k)
        m = fmaxf(m, (float)P3[(p * KNN + k) * 256 + c] * sc + sh);
    A4[p * 256 + c] = (_Float16)fmaxf(m, 0.0f);   // relu(max) == max(relu)
}

__global__ __launch_bounds__(256) void bn_relu_maxn(
    const _Float16* __restrict__ P4, const float* __restrict__ scale,
    const float* __restrict__ shift, float* __restrict__ out)
{
    const int c = threadIdx.x;
    const int b = blockIdx.x;
    const float sc = scale[c], sh = shift[c];
    float m = -3.0e38f;
    for (int n = 0; n < NPTS; ++n)
        m = fmaxf(m, (float)P4[((size_t)b * NPTS + n) * 256 + c] * sc + sh);
    out[b * 256 + c] = fmaxf(m, 0.0f);
}

// ---------------------------------------------------------------------------
// WMMA GEMM + fused BN-stat epilogue.
//   P[M,Nc] = A[M,Kc] @ W[Nc,Kc]^T + bias     (f16 in, f32 accum, f16 out)
// Block = 128 threads (4 waves), tile 64x64, K-step 32.
// Double-buffered LDS tiles filled with GLOBAL_LOAD_ASYNC_TO_LDS_B128;
// s_wait_asynccnt 0x4 releases the current buffer while the next is in flight.
// LDS row stride 40 halves (80 B) keeps 16 B alignment, staggers banks.
// ---------------------------------------------------------------------------
__global__ __launch_bounds__(128) void gemm_wmma_stats(
    const _Float16* __restrict__ A, const _Float16* __restrict__ W,
    const float* __restrict__ bias, _Float16* __restrict__ P,
    float* __restrict__ gSum, float* __restrict__ gSq,
    int M, int Nc, int Kc)
{
    __shared__ __attribute__((aligned(16))) _Float16 As[2][64][40];
    __shared__ __attribute__((aligned(16))) _Float16 Ws[2][64][40];
    __shared__ float sSum[64];
    __shared__ float sSq[64];

    const int tid  = threadIdx.x;
    const int wav  = tid >> 5;
    const int lane = tid & 31;
    const int hlf  = lane >> 4;     // 0: lanes 0-15, 1: lanes 16-31
    const int r    = lane & 15;
    const int m0   = blockIdx.x * 64;
    const int n0   = blockIdx.y * 64;

    if (tid < 64) { sSum[tid] = 0.0f; sSq[tid] = 0.0f; }

    v8f acc[4] = {};

    // Each thread copies one 16-half row-segment of A and of W per K-step.
    const int ldRow = tid >> 1;           // 0..63
    const int ldCol = (tid & 1) * 16;     // 0 or 16 halves
    const unsigned aOffBase = (unsigned)(((m0 + ldRow) * Kc + ldCol) * 2);
    const unsigned wOffBase = (unsigned)(((n0 + ldRow) * Kc + ldCol) * 2);
    const unsigned ldsA0 = (unsigned)(uintptr_t)&As[0][ldRow][ldCol];
    const unsigned ldsA1 = (unsigned)(uintptr_t)&As[1][ldRow][ldCol];
    const unsigned ldsW0 = (unsigned)(uintptr_t)&Ws[0][ldRow][ldCol];
    const unsigned ldsW1 = (unsigned)(uintptr_t)&Ws[1][ldRow][ldCol];

    const int S = Kc >> 5;                // K-steps (2 / 4 / 8 here)

    // prologue: fill buffer 0
    {
        const unsigned ao = aOffBase, wo = wOffBase;
        ASYNC_CP16(ldsA0,       ao,       A);
        ASYNC_CP16(ldsA0 + 16u, ao + 16u, A);
        ASYNC_CP16(ldsW0,       wo,       W);
        ASYNC_CP16(ldsW0 + 16u, wo + 16u, W);
    }

    for (int s = 0; s < S; ++s) {
        const int cur = s & 1;
        if (s + 1 < S) {                  // issue next slice into other buffer
            const unsigned ao = aOffBase + (unsigned)((s + 1) << 6);  // 32 halves = 64 B
            const unsigned wo = wOffBase + (unsigned)((s + 1) << 6);
            const unsigned la = cur ? ldsA0 : ldsA1;
            const unsigned lw = cur ? ldsW0 : ldsW1;
            ASYNC_CP16(la,       ao,       A);
            ASYNC_CP16(la + 16u, ao + 16u, A);
            ASYNC_CP16(lw,       wo,       W);
            ASYNC_CP16(lw + 16u, wo + 16u, W);
            asm volatile("s_wait_asynccnt 0x4" ::: "memory");  // current buf done
        } else {
            asm volatile("s_wait_asynccnt 0x0" ::: "memory");  // all done
        }
        __syncthreads();                  // everyone's async writes visible

        // A fragment (16x32 f16): lane<16 -> K0-7|K16-23, lane>=16 -> K8-15|K24-31
        const h8 alo = *(const h8*)&As[cur][wav * 16 + r][8 * hlf];
        const h8 ahi = *(const h8*)&As[cur][wav * 16 + r][16 + 8 * hlf];
        const v16h af = __builtin_shufflevector(alo, ahi,
            0,1,2,3,4,5,6,7,8,9,10,11,12,13,14,15);

        // B fragments (32x16): lane<16 -> K0-7|K16-23, lane>=16 -> K8-15|K24-31
        #pragma unroll
        for (int t = 0; t < 4; ++t) {
            const h8 blo = *(const h8*)&Ws[cur][t * 16 + r][8 * hlf];
            const h8 bhi = *(const h8*)&Ws[cur][t * 16 + r][16 + 8 * hlf];
            const v16h bf = __builtin_shufflevector(blo, bhi,
                0,1,2,3,4,5,6,7,8,9,10,11,12,13,14,15);
            acc[t] = __builtin_amdgcn_wmma_f32_16x16x32_f16(
                false, af, false, bf, (short)0, acc[t], false, false);
        }
        __syncthreads();                  // readers done before buffer reuse
    }

    // Epilogue: +bias, store f16 pre-activation, fused per-channel stats.
    // D layout: vgpr e, lane -> (row = e + 8*hlf, col = r) within 16x16 tile.
    #pragma unroll
    for (int t = 0; t < 4; ++t) {
        const int col = n0 + t * 16 + r;
        const float bc = bias[col];
        float s = 0.0f, q = 0.0f;
        #pragma unroll
        for (int e = 0; e < 8; ++e) {
            const int row = m0 + wav * 16 + hlf * 8 + e;
            const float v = acc[t][e] + bc;
            P[(size_t)row * Nc + col] = (_Float16)v;
            s += v; q += v * v;
        }
        atomicAdd(&sSum[t * 16 + r], s);   // ds_add_f32
        atomicAdd(&sSq[t * 16 + r], q);
    }
    __syncthreads();
    if (tid < 64) {
        atomicAdd(&gSum[n0 + tid], sSum[tid]);
        atomicAdd(&gSq[n0 + tid], sSq[tid]);
    }
}

// ---------------------------------------------------------------------------
// Small helpers
// ---------------------------------------------------------------------------
__global__ void zero_stats(float* __restrict__ s)
{
    const int tid = threadIdx.x;
    for (int i = 0; i < 16; ++i) s[tid + i * 256] = 0.0f;   // 4096 floats
}

__global__ void cvt_weights(const float* __restrict__ w2, const float* __restrict__ w3,
                            const float* __restrict__ wp, _Float16* __restrict__ w2h,
                            _Float16* __restrict__ w3h, _Float16* __restrict__ wph)
{
    const int i = blockIdx.x * 256 + threadIdx.x;   // 65536 threads
    if (i < 128 * 64)  w2h[i] = (_Float16)w2[i];
    if (i < 256 * 128) w3h[i] = (_Float16)w3[i];
    wph[i] = (_Float16)wp[i];                        // 256*256 == grid size
}

// ---------------------------------------------------------------------------
// Workspace layout (bytes)
// ---------------------------------------------------------------------------
#define OFF_STATS 0u                                   // 4 layers * 1024 f32 = 16 KB
#define OFF_W2H   16384u
#define OFF_W3H   32768u                               // w2h: 8192 h = 16 KB
#define OFF_WPH   98304u                               // w3h: 32768 h = 64 KB
#define OFF_P1    229376u                              // wph: 65536 h = 128 KB
#define OFF_P2    (OFF_P1 + (size_t)NEDGE * 64 * 2)    // P1: 84 MB
#define OFF_P3    (OFF_P2 + (size_t)NEDGE * 128 * 2)   // P2: 168 MB
#define OFF_A4    (OFF_P3 + (size_t)NEDGE * 256 * 2)   // P3: 336 MB
#define OFF_P4    (OFF_A4 + (size_t)NPOINT * 256 * 2)  // A4: 16.8 MB

extern "C" void kernel_launch(void* const* d_in, const int* in_sizes, int n_in,
                              void* d_out, int out_size, void* d_ws, size_t ws_size,
                              hipStream_t stream) {
    const float* x   = (const float*)d_in[0];
    const float* w1  = (const float*)d_in[1];
    const float* b1  = (const float*)d_in[2];
    const float* g1  = (const float*)d_in[3];
    const float* be1 = (const float*)d_in[4];
    const float* w2  = (const float*)d_in[5];
    const float* b2  = (const float*)d_in[6];
    const float* g2  = (const float*)d_in[7];
    const float* be2 = (const float*)d_in[8];
    const float* w3  = (const float*)d_in[9];
    const float* b3  = (const float*)d_in[10];
    const float* g3  = (const float*)d_in[11];
    const float* be3 = (const float*)d_in[12];
    const float* wp  = (const float*)d_in[13];
    const float* bp  = (const float*)d_in[14];
    const float* gp  = (const float*)d_in[15];
    const float* bep = (const float*)d_in[16];

    char* ws = (char*)d_ws;
    float*    stats = (float*)(ws + OFF_STATS);
    _Float16* w2h   = (_Float16*)(ws + OFF_W2H);
    _Float16* w3h   = (_Float16*)(ws + OFF_W3H);
    _Float16* wph   = (_Float16*)(ws + OFF_WPH);
    _Float16* P1    = (_Float16*)(ws + OFF_P1);
    _Float16* P2    = (_Float16*)(ws + OFF_P2);
    _Float16* P3    = (_Float16*)(ws + OFF_P3);
    _Float16* A4    = (_Float16*)(ws + OFF_A4);
    _Float16* P4    = (_Float16*)(ws + OFF_P4);
    float* out = (float*)d_out;

    zero_stats<<<1, 256, 0, stream>>>(stats);
    cvt_weights<<<256, 256, 0, stream>>>(w2, w3, wp, w2h, w3h, wph);

    // layer 1: KNN + edge + 6->64 (VALU, LDS-resident point cloud)
    knn_edge_l1<<<dim3(NPTS / 256, NB), 256, 0, stream>>>(x, w1, b1, P1);
    col_stats<<<1024, 256, 0, stream>>>(P1, stats + 0, stats + 256, NEDGE, 64);
    finalize_bn<<<1, 256, 0, stream>>>(stats + 0, g1, be1, (float)NEDGE, 64);
    bn_relu<<<4096, 256, 0, stream>>>(P1, stats + 512, stats + 768, 63,
                                      (size_t)NEDGE * 64);

    // layer 2: 64 -> 128 (WMMA)
    gemm_wmma_stats<<<dim3(NEDGE / 64, 128 / 64), 128, 0, stream>>>(
        P1, w2h, b2, P2, stats + 1024, stats + 1280, NEDGE, 128, 64);
    finalize_bn<<<1, 256, 0, stream>>>(stats + 1024, g2, be2, (float)NEDGE, 128);
    bn_relu<<<4096, 256, 0, stream>>>(P2, stats + 1536, stats + 1792, 127,
                                      (size_t)NEDGE * 128);

    // layer 3: 128 -> 256 (WMMA), then BN+ReLU fused with max over K
    gemm_wmma_stats<<<dim3(NEDGE / 64, 256 / 64), 128, 0, stream>>>(
        P2, w3h, b3, P3, stats + 2048, stats + 2304, NEDGE, 256, 128);
    finalize_bn<<<1, 256, 0, stream>>>(stats + 2048, g3, be3, (float)NEDGE, 256);
    bn_relu_maxk<<<NPOINT, 256, 0, stream>>>(P3, stats + 2560, stats + 2816, A4);

    // final: 256 -> 256 (WMMA), BN+ReLU fused with max over N
    gemm_wmma_stats<<<dim3(NPOINT / 64, 256 / 64), 128, 0, stream>>>(
        A4, wph, bp, P4, stats + 3072, stats + 3328, NPOINT, 256, 256);
    finalize_bn<<<1, 256, 0, stream>>>(stats + 3072, gp, bep, (float)NPOINT, 256);
    bn_relu_maxn<<<8, 256, 0, stream>>>(P4, stats + 3584, stats + 3840, out);
}
